// TGCNCell_19816979104199
// MI455X (gfx1250) — matrix operand order repaired and checked by reading, not compiled, not run on hardware
//
#include <hip/hip_runtime.h>

// ---------------------------------------------------------------------------
// T-GCN cell for MI455X (gfx1250, wave32, WMMA + async-to-LDS staging).
//  Per step t:
//   1) k_pgates: PgT[col][n]  = feats(b,n,:) . Wg[o,:]      (col = b*128+o), bf16
//   2) k_gemm<0>: gates[m][col] = sigmoid( sum_n Lbf[m][n]*PgT[col][n] + bg[o] )
//   3) k_qcand : QcT[col][n]  = feats2(b,n,:) . Wc[h,:]     (col = b*64+h), bf16
//   4) k_gemm<1>: cand + fused GRU update -> new state + outs[t]
//  GEMM: v_wmma_f32_16x16x32_bf16, A/B tiles staged to LDS with
//  global_load_async_to_lds_b128 (ASYNCcnt), double buffered, K=64 per stage.
// ---------------------------------------------------------------------------

typedef __bf16 bf16_t;
typedef __attribute__((ext_vector_type(16))) __bf16 v16bf;
typedef __attribute__((ext_vector_type(8)))  float  v8f;

#define N_NODES 2048
#define HID     64
#define BATCH   8
#define SEQ     32
#define NH      (N_NODES * HID)     // 131072
#define GCOLS   (BATCH * 2 * HID)   // 1024 (gates GEMM N)
#define CCOLS   (BATCH * HID)       // 512  (cand GEMM N)

#define KCH       64                 // K per stage (two WMMA K-steps)
#define LDS_STR   72                 // row stride in bf16 (64 + 8 pad, 144B, 16B-aligned)
#define BUF_ELEMS (128 * LDS_STR)    // elements per buffer (9216)
#define BUF_BYTES (BUF_ELEMS * 2)    // 18432 B   (total LDS: 4 buffers = 72 KB)

// ---------------- WMMA + fragment helpers ----------------------------------
__device__ __forceinline__ v8f wmma_bf16(v16bf a, v16bf b, v8f c) {
    return __builtin_amdgcn_wmma_f32_16x16x32_bf16(
        false, a, false, b, (short)0, c, false, false);
}
// A operand 16x32 bf16 from LDS (pointer pre-offset row*LDS_STR + half*8 + kk):
//   elements +0..7 and +16..23
__device__ __forceinline__ v16bf frag_a(const bf16_t* p) {
    union { uint4 u[2]; v16bf v; } t;
    t.u[0] = *(const uint4*)(p);
    t.u[1] = *(const uint4*)(p + 16);
    return t.v;
}
// B operand 32x16 bf16 from LDS (pointer pre-offset row*LDS_STR + half*8 + kk):
//   elements +0..7 and +16..23
__device__ __forceinline__ v16bf frag_b(const bf16_t* p) {
    union { uint4 u[2]; v16bf v; } t;
    t.u[0] = *(const uint4*)(p);
    t.u[1] = *(const uint4*)(p + 16);
    return t.v;
}
// gfx1250 async copy: LDS[lds_byte_off] = MEM[gptr], 16 bytes, tracked by ASYNCcnt
__device__ __forceinline__ void async_b128(unsigned lds_byte_off, const bf16_t* gptr) {
    asm volatile("global_load_async_to_lds_b128 %0, %1, off"
                 :: "v"(lds_byte_off), "v"(gptr) : "memory");
}
__device__ __forceinline__ void wait_async0() {
    asm volatile("s_wait_asynccnt 0" ::: "memory");
}

// ---------------- init: L fp32 -> bf16, state = 0 --------------------------
__global__ void k_init(const float* __restrict__ L, bf16_t* __restrict__ Lbf,
                       float* __restrict__ state) {
    int i = blockIdx.x * 256 + threadIdx.x;
    if (i < N_NODES * N_NODES) Lbf[i] = (bf16_t)L[i];
    if (i < BATCH * NH)        state[i] = 0.0f;
}

// ---------------- PgT[col][n] = feats . Wg row (8 outputs/thread) ----------
__global__ __launch_bounds__(256)
void k_pgates(const float* __restrict__ x, int t,
              const float* __restrict__ state,
              const float* __restrict__ Wg,
              bf16_t* __restrict__ PgT) {
    int idx  = blockIdx.x * 256 + threadIdx.x;       // (GCOLS/8)*2048 threads
    int n    = idx & (N_NODES - 1);
    int col8 = idx >> 11;                            // 0..127
    int b    = col8 >> 4;
    int o0   = (col8 & 15) << 3;
    const float* s = state + (size_t)b * NH + (size_t)n * HID;
    float xv = x[((size_t)b * SEQ + t) * N_NODES + n];
    float acc[8];
    const float* w[8];
    #pragma unroll
    for (int i = 0; i < 8; i++) {
        w[i] = Wg + (size_t)(o0 + i) * 65;
        acc[i] = w[i][0] * xv;
    }
    #pragma unroll 4
    for (int j = 0; j < HID; j++) {
        float sj = s[j];
        #pragma unroll
        for (int i = 0; i < 8; i++) acc[i] += w[i][1 + j] * sj;
    }
    size_t base = (size_t)(col8 * 8) * N_NODES + n;
    #pragma unroll
    for (int i = 0; i < 8; i++) PgT[base + (size_t)i * N_NODES] = (bf16_t)acc[i];
}

// ---------------- QcT[col][n] = [x, r*state] . Wc row ----------------------
// r follows the reference's FLAT split: r_flat[b][i] = gates[b][i>>7][i&127]
__global__ __launch_bounds__(256)
void k_qcand(const float* __restrict__ x, int t,
             const float* __restrict__ state,
             const float* __restrict__ gates,
             const float* __restrict__ Wc,
             bf16_t* __restrict__ QcT) {
    int idx  = blockIdx.x * 256 + threadIdx.x;       // (CCOLS/8)*2048 threads
    int n    = idx & (N_NODES - 1);
    int col8 = idx >> 11;                            // 0..63
    int b    = col8 >> 3;
    int h0   = (col8 & 7) << 3;
    const float* s = state + (size_t)b * NH + (size_t)n * HID;
    const float* g = gates + (size_t)(n >> 1) * GCOLS + b * 128 + (n & 1) * 64;
    float xv = x[((size_t)b * SEQ + t) * N_NODES + n];
    float acc[8];
    const float* w[8];
    #pragma unroll
    for (int i = 0; i < 8; i++) {
        w[i] = Wc + (size_t)(h0 + i) * 65;
        acc[i] = w[i][0] * xv;
    }
    #pragma unroll 4
    for (int j = 0; j < HID; j++) {
        float rs = g[j] * s[j];
        #pragma unroll
        for (int i = 0; i < 8; i++) acc[i] += w[i][1 + j] * rs;
    }
    size_t base = (size_t)(col8 * 8) * N_NODES + n;
    #pragma unroll
    for (int i = 0; i < 8; i++) QcT[base + (size_t)i * N_NODES] = (bf16_t)acc[i];
}

// ---------------- big GEMM: D[m][col] = sum_n A[m][n] * BT[col][n] ---------
// Block tile 128x128, 8 waves (4x2 wave tiles of 32x64), 2x4 accumulators
// of 16x16 per wave.  K staged to LDS in chunks of 64 via async copies,
// double buffered; 16 v_wmma per stage; branchless mainloop (last stage peeled).
template <int EPI>
__global__ __launch_bounds__(256)
void k_gemm(const bf16_t* __restrict__ A, const bf16_t* __restrict__ BT,
            const float* __restrict__ bias,
            float* __restrict__ gates,
            float* __restrict__ state,
            float* __restrict__ out_t) {
    __shared__ bf16_t shA[2 * BUF_ELEMS];
    __shared__ bf16_t shB[2 * BUF_ELEMS];

    const int tid  = threadIdx.x;
    const int lane = tid & 31;
    const int wv   = tid >> 5;
    const int wm   = wv >> 1;                 // 0..3  (32-row stripes)
    const int wn   = wv & 1;                  // 0..1  (64-col stripes)
    const int mblk = blockIdx.y * 128;
    const int cblk = blockIdx.x * 128;
    const int r    = lane & 15;
    const int half = lane >> 4;

    // --- staging map: 256 threads cover 128 rows x 64 K of A and of B ------
    const int seg  = tid & 7;                 // 16B segment within 64-elem row
    const int crow = tid >> 3;                // 0..31 (rows crow+{0,32,64,96})
    const bf16_t* gA = A  + (size_t)(mblk + crow) * N_NODES + seg * 8;
    const bf16_t* gB = BT + (size_t)(cblk + crow) * N_NODES + seg * 8;
    const unsigned aOff = (unsigned)(uintptr_t)shA + (unsigned)(crow * LDS_STR + seg * 8) * 2;
    const unsigned bOff = (unsigned)(uintptr_t)shB + (unsigned)(crow * LDS_STR + seg * 8) * 2;
    const unsigned rowJ = (unsigned)(32 * LDS_STR * 2);      // +32 rows in LDS
    const size_t   gJ   = (size_t)32 * N_NODES;              // +32 rows in global

    // per-wave fragment offsets within a buffer (elements)
    const int aF0 = (wm * 32 +      r) * LDS_STR + half * 8;
    const int aF1 = (wm * 32 + 16 + r) * LDS_STR + half * 8;
    const int bF0 = (wn * 64 +      r) * LDS_STR + half * 8;
    const int bF1 = (wn * 64 + 16 + r) * LDS_STR + half * 8;
    const int bF2 = (wn * 64 + 32 + r) * LDS_STR + half * 8;
    const int bF3 = (wn * 64 + 48 + r) * LDS_STR + half * 8;

    v8f acc[2][4];
    #pragma unroll
    for (int i = 0; i < 2; i++)
        #pragma unroll
        for (int j = 0; j < 4; j++)
            acc[i][j] = (v8f){0.f, 0.f, 0.f, 0.f, 0.f, 0.f, 0.f, 0.f};

    #define STAGE(BUFO, K)                                                   \
        do {                                                                 \
            _Pragma("unroll")                                                \
            for (int q = 0; q < 4; q++) {                                    \
                async_b128(aOff + (BUFO) + q * rowJ, gA + q * gJ + (K));     \
                async_b128(bOff + (BUFO) + q * rowJ, gB + q * gJ + (K));     \
            }                                                                \
        } while (0)

    #define COMPUTE(BUF)                                                     \
        do {                                                                 \
            const bf16_t* As = shA + (BUF) * BUF_ELEMS;                      \
            const bf16_t* Bs = shB + (BUF) * BUF_ELEMS;                      \
            _Pragma("unroll")                                                \
            for (int kk = 0; kk < KCH; kk += 32) {                           \
                v16bf af0 = frag_a(As + aF0 + kk);                           \
                v16bf af1 = frag_a(As + aF1 + kk);                           \
                v16bf bq0 = frag_b(Bs + bF0 + kk);                           \
                v16bf bq1 = frag_b(Bs + bF1 + kk);                           \
                v16bf bq2 = frag_b(Bs + bF2 + kk);                           \
                v16bf bq3 = frag_b(Bs + bF3 + kk);                           \
                acc[0][0] = wmma_bf16(af0, bq0, acc[0][0]);                  \
                acc[0][1] = wmma_bf16(af0, bq1, acc[0][1]);                  \
                acc[0][2] = wmma_bf16(af0, bq2, acc[0][2]);                  \
                acc[0][3] = wmma_bf16(af0, bq3, acc[0][3]);                  \
                acc[1][0] = wmma_bf16(af1, bq0, acc[1][0]);                  \
                acc[1][1] = wmma_bf16(af1, bq1, acc[1][1]);                  \
                acc[1][2] = wmma_bf16(af1, bq2, acc[1][2]);                  \
                acc[1][3] = wmma_bf16(af1, bq3, acc[1][3]);                  \
            }                                                                \
        } while (0)

    const int NS = N_NODES / KCH;             // 32 stages
    STAGE(0u, 0);                             // prologue: stage 0 -> buffer 0
    #pragma unroll 2
    for (int ks = 0; ks < NS - 1; ks++) {     // branchless: always stage next
        const int cur = ks & 1;
        wait_async0();                        // our copies for stage ks landed
        __syncthreads();                      // all copies landed; prev reads done
        STAGE((unsigned)((cur ^ 1) * BUF_BYTES), (ks + 1) * KCH);
        COMPUTE(cur);
    }
    wait_async0();                            // final stage (peeled, no staging)
    __syncthreads();
    COMPUTE((NS - 1) & 1);

    #undef STAGE
    #undef COMPUTE

    // epilogue: element j of v8f in lane l is D[m0 + (l>>4)*8 + j][c0 + (l&15)]
    #pragma unroll
    for (int tm = 0; tm < 2; tm++) {
        #pragma unroll
        for (int tn = 0; tn < 4; tn++) {
            #pragma unroll
            for (int j = 0; j < 8; j++) {
                int m = mblk + wm * 32 + tm * 16 + half * 8 + j;
                int c = cblk + wn * 64 + tn * 16 + r;
                float v = acc[tm][tn][j];
                if (EPI == 0) {
                    int o = c & 127;
                    v += bias[o];
                    float gate = 1.0f / (1.0f + __expf(-v));
                    gates[(size_t)m * GCOLS + c] = gate;
                } else {
                    int h = c & 63;
                    int b = c >> 6;
                    v += bias[h];
                    float cd = tanhf(v);
                    int i = m * HID + h;   // flat (n,h) index within batch
                    float u = gates[(size_t)(1024 + (m >> 1)) * GCOLS
                                    + b * 128 + (m & 1) * 64 + h];
                    float sOld = state[(size_t)b * NH + i];
                    float nh = u * sOld + (1.0f - u) * cd;
                    state[(size_t)b * NH + i] = nh;
                    out_t[(size_t)b * NH + i] = nh;
                }
            }
        }
    }
}

// ---------------------------------------------------------------------------
extern "C" void kernel_launch(void* const* d_in, const int* in_sizes, int n_in,
                              void* d_out, int out_size, void* d_ws, size_t ws_size,
                              hipStream_t stream) {
    const float* x  = (const float*)d_in[0];   // (B, S, N)
    const float* L  = (const float*)d_in[1];   // (N, N)
    const float* Wg = (const float*)d_in[2];   // (128, 65)
    const float* bg = (const float*)d_in[3];   // (128,)
    const float* Wc = (const float*)d_in[4];   // (64, 65)
    const float* bc = (const float*)d_in[5];   // (64,)
    float* out = (float*)d_out;                // (S, B, N*H)

    char* ws = (char*)d_ws;
    bf16_t* Lbf   = (bf16_t*)(ws);                      //  8 MiB
    float*  state = (float*) (ws + ( 8u << 20));        //  4 MiB
    float*  gates = (float*) (ws + (12u << 20));        //  8 MiB
    bf16_t* PgT   = (bf16_t*)(ws + (20u << 20));        //  4 MiB
    bf16_t* QcT   = (bf16_t*)(ws + (24u << 20));        //  2 MiB  (26 MiB total)

    k_init<<<(N_NODES * N_NODES + 255) / 256, 256, 0, stream>>>(L, Lbf, state);

    for (int t = 0; t < SEQ; t++) {
        k_pgates<<<(GCOLS / 8) * N_NODES / 256, 256, 0, stream>>>(x, t, state, Wg, PgT);

        dim3 gGates(GCOLS / 128, N_NODES / 128);   // (8, 16)
        k_gemm<0><<<gGates, 256, 0, stream>>>(Lbf, PgT, bg, gates, nullptr, nullptr);

        k_qcand<<<(CCOLS / 8) * N_NODES / 256, 256, 0, stream>>>(x, t, state, gates, Wc, QcT);

        dim3 gCand(CCOLS / 128, N_NODES / 128);    // (4, 16)
        k_gemm<1><<<gCand, 256, 0, stream>>>(Lbf, QcT, bc, gates, state,
                                             out + (size_t)t * BATCH * NH);
    }
}
